// Decoder_27006754357834
// MI455X (gfx1250) — compile-verified, matrix operand order checked by reading, not run on hardware
//
#include <hip/hip_runtime.h>
#include <hip/hip_bf16.h>
#include <math.h>

#define B_ 128
#define S_ 512
#define H_ 1024
#define E_ 1024
#define V_ 50000
#define EXTRA_ 500
#define EPS_ 1e-8f

typedef __attribute__((ext_vector_type(16))) __bf16 v16bf;
typedef __attribute__((ext_vector_type(8)))  float  v8f;
typedef unsigned int v4u __attribute__((ext_vector_type(4)));
typedef int          v8i __attribute__((ext_vector_type(8)));
typedef int          v4i __attribute__((ext_vector_type(4)));

union BF16Frag { v16bf v; uint4 q[2]; unsigned int u[8]; };

// A/B 16-bit fragment from bf16 LDS tile [rows][32]:
// lane l: K groups g*16 + (l>>4)*8 + {0..7} -> two 16B ds loads
__device__ __forceinline__ void frag_from_lds_bf16(BF16Frag& f, const __bf16* lds,
                                                   int row, int khalf) {
  const __bf16* p = lds + row * 32 + khalf * 8;
  f.q[0] = *(const uint4*)(p);
  f.q[1] = *(const uint4*)(p + 16);
}

// Same fragment sourced from an fp32 LDS tile (TDM-staged); native f32->bf16 converts.
__device__ __forceinline__ void frag_from_lds_f32(BF16Frag& f, const float* lds,
                                                  int row, int khalf) {
  const float* p = lds + row * 32 + khalf * 8;
  float4 a = *(const float4*)(p);
  float4 b = *(const float4*)(p + 4);
  float4 c = *(const float4*)(p + 16);
  float4 d = *(const float4*)(p + 20);
  f.v[0]  = (__bf16)a.x; f.v[1]  = (__bf16)a.y; f.v[2]  = (__bf16)a.z; f.v[3]  = (__bf16)a.w;
  f.v[4]  = (__bf16)b.x; f.v[5]  = (__bf16)b.y; f.v[6]  = (__bf16)b.z; f.v[7]  = (__bf16)b.w;
  f.v[8]  = (__bf16)c.x; f.v[9]  = (__bf16)c.y; f.v[10] = (__bf16)c.z; f.v[11] = (__bf16)c.w;
  f.v[12] = (__bf16)d.x; f.v[13] = (__bf16)d.y; f.v[14] = (__bf16)d.z; f.v[15] = (__bf16)d.w;
}

// ---- Tensor Data Mover: issue async DMA of a [128 rows x 32 cols] fp32 tile
// (row stride = ld elements) from global memory into LDS (contiguous, row-major).
// D# layout per CDNA5 ISA.
__device__ __forceinline__ void tdm_load_tile_128x32(const float* gtile, unsigned lds_off,
                                                     unsigned ld_elems) {
  unsigned long long ga = (unsigned long long)(uintptr_t)gtile;
  v4u g0;
  g0[0] = 1u;                                          // count=1, user descriptor
  g0[1] = lds_off;                                     // LDS byte address
  g0[2] = (unsigned)(ga & 0xFFFFFFFFu);                // global_addr[31:0]
  g0[3] = (unsigned)((ga >> 32) & 0x1FFFFFFu) | (2u << 30);  // addr[56:32] | type=2
  const unsigned TD0 = 1u << 20, TD1 = 1u << 20;       // generous tensor dims (tile in-bounds)
  v8i g1;
  g1[0] = (int)0x00020000u;                            // wg_mask=0, data_size=2 (4B)
  g1[1] = (int)((TD0 & 0xFFFFu) << 16);                // tensor_dim0[15:0]
  g1[2] = (int)(((TD0 >> 16) & 0xFFFFu) | ((TD1 & 0xFFFFu) << 16));
  g1[3] = (int)(((TD1 >> 16) & 0xFFFFu) | (32u << 16));   // tile_dim0 = 32
  g1[4] = (int)128u;                                   // tile_dim1 = 128, tile_dim2 = 0
  g1[5] = (int)ld_elems;                               // tensor_dim0_stride (elements)
  g1[6] = 0;
  g1[7] = 0;
  v4i z4 = {0, 0, 0, 0};
#if defined(__clang_major__) && (__clang_major__ >= 23)
  v8i z8 = {0, 0, 0, 0, 0, 0, 0, 0};
  __builtin_amdgcn_tensor_load_to_lds(g0, g1, z4, z4, z8, 0);
#else
  __builtin_amdgcn_tensor_load_to_lds(g0, g1, z4, z4, 0);
#endif
}

// Cooperative stage: 128 rows x 32 K of fp32 -> bf16 LDS tile, 256 threads
__device__ __forceinline__ void stage_tile(__bf16* dst, const float* __restrict__ src,
                                           long long row0, long long ld, int kt) {
  int r = (threadIdx.x >> 3);
  int c = (threadIdx.x & 7) * 4;
  #pragma unroll
  for (int rr = 0; rr < 128; rr += 32) {
    const float* p = src + (row0 + rr + r) * ld + kt + c;
    float4 v = *(const float4*)p;
    __bf16* d = dst + (rr + r) * 32 + c;
    d[0] = (__bf16)v.x; d[1] = (__bf16)v.y; d[2] = (__bf16)v.z; d[3] = (__bf16)v.w;
  }
}

// ---------------- reductions ----------------
__device__ __forceinline__ float block_sum256(float v, float* sm) {
  int t = threadIdx.x; sm[t] = v; __syncthreads();
  #pragma unroll
  for (int s = 128; s > 0; s >>= 1) { if (t < s) sm[t] += sm[t + s]; __syncthreads(); }
  float r = sm[0]; __syncthreads(); return r;
}
__device__ __forceinline__ float block_max256(float v, float* sm) {
  int t = threadIdx.x; sm[t] = v; __syncthreads();
  #pragma unroll
  for (int s = 128; s > 0; s >>= 1) { if (t < s) sm[t] = fmaxf(sm[t], sm[t + s]); __syncthreads(); }
  float r = sm[0]; __syncthreads(); return r;
}

// ---------------- K1: embedding gather + layernorm ----------------
__global__ __launch_bounds__(256) void k_embed_ln(const int* __restrict__ inputs,
    const float* __restrict__ embed, const float* __restrict__ g,
    const float* __restrict__ bta, float* __restrict__ xln) {
  __shared__ float sm[256];
  int b = blockIdx.x, tid = threadIdx.x;
  const float* row = embed + (long long)inputs[b] * E_;
  float s = 0.f;
  for (int i = tid; i < E_; i += 256) s += row[i];
  float mean = block_sum256(s, sm) * (1.f / E_);
  float sq = 0.f;
  for (int i = tid; i < E_; i += 256) { float d = row[i] - mean; sq += d * d; }
  float var = block_sum256(sq, sm) * (1.f / E_);
  float inv = rsqrtf(var + 1e-5f);
  for (int i = tid; i < E_; i += 256)
    xln[(long long)b * E_ + i] = (row[i] - mean) * inv * g[i] + bta[i];
}

// ---------------- generic dot kernels (small GEMMs) ----------------
__global__ __launch_bounds__(256) void k_dot(const float* __restrict__ X,
    const float* __restrict__ W, const float* __restrict__ bias,
    float* __restrict__ out, int N, int K) {
  long long idx = (long long)blockIdx.x * 256 + threadIdx.x;
  int m = (int)(idx / N), n = (int)(idx % N);
  const float4* x = (const float4*)(X + (long long)m * K);
  const float4* w = (const float4*)(W + (long long)n * K);
  float s = 0.f;
  for (int k = 0; k < K / 4; ++k) {
    float4 a = x[k], b = w[k];
    s += a.x * b.x + a.y * b.y + a.z * b.z + a.w * b.w;
  }
  out[idx] = s + bias[n];
}

__global__ __launch_bounds__(256) void k_dot_cat2(const float* __restrict__ X1,
    const float* __restrict__ X2, const float* __restrict__ W,
    const float* __restrict__ bias, float* __restrict__ out, int N, int K1, int K2) {
  long long idx = (long long)blockIdx.x * 256 + threadIdx.x;
  int m = (int)(idx / N), n = (int)(idx % N);
  const float4* x1 = (const float4*)(X1 + (long long)m * K1);
  const float4* x2 = (const float4*)(X2 + (long long)m * K2);
  const float4* w  = (const float4*)(W + (long long)n * (K1 + K2));
  float s = 0.f;
  for (int k = 0; k < K1 / 4; ++k) {
    float4 a = x1[k], b = w[k];
    s += a.x * b.x + a.y * b.y + a.z * b.z + a.w * b.w;
  }
  for (int k = 0; k < K2 / 4; ++k) {
    float4 a = x2[k], b = w[K1 / 4 + k];
    s += a.x * b.x + a.y * b.y + a.z * b.z + a.w * b.w;
  }
  out[idx] = s + bias[n];
}

// ---------------- GRU combine (torch gate order r,z,n) ----------------
__global__ __launch_bounds__(256) void k_gru(const float* __restrict__ gx,
    const float* __restrict__ gh, const float* __restrict__ h0,
    float* __restrict__ s_t, float* __restrict__ s_t_out) {
  int idx = blockIdx.x * 256 + threadIdx.x;      // B*H
  int b = idx / H_, i = idx % H_;
  const float* gxb = gx + (long long)b * 3 * H_;
  const float* ghb = gh + (long long)b * 3 * H_;
  float r = 1.f / (1.f + expf(-(gxb[i] + ghb[i])));
  float z = 1.f / (1.f + expf(-(gxb[H_ + i] + ghb[H_ + i])));
  float nn = tanhf(gxb[2 * H_ + i] + r * ghb[2 * H_ + i]);
  float h = h0[idx];
  float s = (1.f - z) * nn + z * h;
  s_t[idx] = s; s_t_out[idx] = s;
}

// ---------------- fused attention-score GEMM (WMMA bf16 + TDM double buffer) ----
// e_t[b,s] += sum_n Vv[n]*tanh( (enc @ Wh.T)[m,n] + dec_feat[b,n] + cov[m]*Wc[n] )
__global__ __launch_bounds__(256) void k_attn_scores(
    const float* __restrict__ enc,       // [B*S, H]
    const float* __restrict__ Wh,        // [H, H]
    const float* __restrict__ dec_feat,  // [B, H]
    const float* __restrict__ coverage,  // [B*S]
    const float* __restrict__ Wc,        // [H]
    const float* __restrict__ Vv,        // [H]
    float* __restrict__ e_t)             // [B*S], pre-zeroed
{
  __shared__ float lA[2][128 * 32];      // fp32 tiles staged by TDM, double-buffered
  __shared__ float lB[2][128 * 32];
  const int m0 = blockIdx.x * 128;       // row block over B*S (128 | 512 -> single b)
  const int n0 = blockIdx.y * 128;       // col block over H
  const int b  = m0 / S_;
  const int lane  = threadIdx.x & 31;
  const int wave  = threadIdx.x >> 5;
  const int wm    = wave & 3;            // rows wm*32 .. +31
  const int wn    = wave >> 2;           // cols wn*64 .. +63
  const int khalf = lane >> 4;
  const int nl    = lane & 15;

  const unsigned ldsA0 = (unsigned)(uintptr_t)(&lA[0][0]);
  const unsigned ldsA1 = (unsigned)(uintptr_t)(&lA[1][0]);
  const unsigned ldsB0 = (unsigned)(uintptr_t)(&lB[0][0]);
  const unsigned ldsB1 = (unsigned)(uintptr_t)(&lB[1][0]);

  v8f acc[2][4];
  #pragma unroll
  for (int a = 0; a < 2; ++a)
    #pragma unroll
    for (int c = 0; c < 4; ++c)
      acc[a][c] = (v8f){0.f, 0.f, 0.f, 0.f, 0.f, 0.f, 0.f, 0.f};

  // prologue: prefetch K-chunk 0 into buffer 0 (one wave drives the TDM)
  if (wave == 0) {
    tdm_load_tile_128x32(enc + (long long)m0 * H_, ldsA0, H_);
    tdm_load_tile_128x32(Wh  + (long long)n0 * H_, ldsB0, H_);
  }

  for (int k = 0; k < H_ / 32; ++k) {
    const int cur = k & 1;
    if (wave == 0) {
      if (k + 1 < H_ / 32) {
        // prefetch next chunk into the other buffer, then wait for current chunk
        const unsigned la = (cur ? ldsA0 : ldsA1);
        const unsigned lb = (cur ? ldsB0 : ldsB1);
        tdm_load_tile_128x32(enc + (long long)m0 * H_ + (k + 1) * 32, la, H_);
        tdm_load_tile_128x32(Wh  + (long long)n0 * H_ + (k + 1) * 32, lb, H_);
        __builtin_amdgcn_s_wait_tensorcnt(2);   // only the 2 newest may remain in flight
      } else {
        __builtin_amdgcn_s_wait_tensorcnt(0);
      }
    }
    __syncthreads();                            // current buffer ready for all waves

    const float* lAc = &lA[cur][0];
    const float* lBc = &lB[cur][0];
    BF16Frag Af[2], Bf[4];
    #pragma unroll
    for (int a = 0; a < 2; ++a) frag_from_lds_f32(Af[a], lAc, wm * 32 + a * 16 + nl, khalf);
    #pragma unroll
    for (int c = 0; c < 4; ++c) frag_from_lds_f32(Bf[c], lBc, wn * 64 + c * 16 + nl, khalf);
    #pragma unroll
    for (int a = 0; a < 2; ++a)
      #pragma unroll
      for (int c = 0; c < 4; ++c)
        acc[a][c] = __builtin_amdgcn_wmma_f32_16x16x32_bf16(
            false, Af[a].v, false, Bf[c].v, (short)0, acc[a][c], false, false);
    __syncthreads();                            // all reads done before buffer is refilled
  }

  // fused epilogue: tanh, dot with Vv, reduce over this block's 128 n-cols
  float df[4], wc[4], vv[4];
  #pragma unroll
  for (int c = 0; c < 4; ++c) {
    int n = n0 + wn * 64 + c * 16 + nl;
    df[c] = dec_feat[(long long)b * H_ + n];
    wc[c] = Wc[n];
    vv[c] = Vv[n];
  }
  #pragma unroll
  for (int a = 0; a < 2; ++a) {
    #pragma unroll
    for (int j = 0; j < 8; ++j) {
      int m = m0 + wm * 32 + a * 16 + j + 8 * khalf;
      float cov = coverage[m];
      float rowsum = 0.f;
      #pragma unroll
      for (int c = 0; c < 4; ++c) {
        float val = acc[a][c][j] + df[c] + cov * wc[c];
        rowsum += vv[c] * tanhf(val);
      }
      rowsum += __shfl_xor(rowsum, 1);
      rowsum += __shfl_xor(rowsum, 2);
      rowsum += __shfl_xor(rowsum, 4);
      rowsum += __shfl_xor(rowsum, 8);     // sum within 16-lane half (same m)
      if (nl == 0) atomicAdd(&e_t[m], rowsum);
    }
  }
}

// ---------------- softmax over S + coverage update/loss ----------------
__global__ __launch_bounds__(256) void k_softmax_cov(const float* __restrict__ e_t,
    const float* __restrict__ mask, const float* __restrict__ coverage,
    float* __restrict__ atten, float* __restrict__ cov_next, float* __restrict__ cov_loss) {
  __shared__ float sm[256];
  int b = blockIdx.x, tid = threadIdx.x;
  const float* e  = e_t  + (long long)b * S_;
  const float* mk = mask + (long long)b * S_;
  float mx = -1e30f;
  for (int s = tid; s < S_; s += 256) mx = fmaxf(mx, e[s] + mk[s]);
  mx = block_max256(mx, sm);
  float sum = 0.f;
  for (int s = tid; s < S_; s += 256) sum += expf(e[s] + mk[s] - mx);
  sum = block_sum256(sum, sm);
  float inv = 1.f / sum;
  float lmin = 0.f;
  for (int s = tid; s < S_; s += 256) {
    float a  = expf(e[s] + mk[s] - mx) * inv;
    float cn = coverage[(long long)b * S_ + s] + a;
    atten[(long long)b * S_ + s]    = a;
    cov_next[(long long)b * S_ + s] = cn;
    lmin += fminf(a, cn);
  }
  lmin = block_sum256(lmin, sm);
  if (tid == 0) cov_loss[b] = lmin;
}

// ---------------- context vector h_t = atten @ enc_outputs ----------------
__global__ __launch_bounds__(256) void k_context(const float* __restrict__ atten,
    const float* __restrict__ enc, float* __restrict__ h_t) {
  __shared__ float sa[S_];
  int idx = blockIdx.x * 256 + threadIdx.x;     // B*H; block spans single b
  int b = idx / H_, h = idx % H_;
  for (int s = threadIdx.x; s < S_; s += 256) sa[s] = atten[(long long)b * S_ + s];
  __syncthreads();
  float acc = 0.f;
  const float* ep = enc + (long long)b * S_ * H_ + h;
  for (int s = 0; s < S_; ++s) acc += sa[s] * ep[(long long)s * H_];
  h_t[idx] = acc;
}

// ---------------- pointer-generator gate ----------------
__global__ __launch_bounds__(256) void k_pgen(const float* __restrict__ h_t,
    const float* __restrict__ s_t, const float* __restrict__ x,
    const float* __restrict__ pg_W, const float* __restrict__ pg_b,
    float* __restrict__ p_gen) {
  __shared__ float sm[256];
  int b = blockIdx.x, tid = threadIdx.x;
  float s = 0.f;
  for (int j = tid; j < 3 * H_; j += 256) {
    float v = (j < H_) ? h_t[(long long)b * H_ + j]
            : (j < 2 * H_) ? s_t[(long long)b * H_ + j - H_]
            : x[(long long)b * H_ + j - 2 * H_];
    s += v * pg_W[j];
  }
  s = block_sum256(s, sm);
  if (tid == 0) {
    float p = 1.f / (1.f + expf(-(s + pg_b[0])));
    p_gen[b] = fmaxf(p, EPS_);
  }
}

// ---------------- out2 GEMM (WMMA bf16): logits = h1 @ out2_W.T + b ----------------
__global__ __launch_bounds__(256) void k_out2_gemm(const float* __restrict__ h1,  // [B,H]
    const float* __restrict__ W2,    // [V, H]
    const float* __restrict__ b2,    // [V]
    float* __restrict__ logits) {    // [B, V]
  __shared__ __bf16 lA[128 * 32];
  const int n0 = blockIdx.x * 128;
  const int lane  = threadIdx.x & 31;
  const int wave  = threadIdx.x >> 5;         // each wave: 16 vocab cols, all 128 batch rows
  const int khalf = lane >> 4;
  const int nl    = lane & 15;
  const int nrow  = n0 + wave * 16 + nl;      // vocab column this lane owns
  const bool valid = (nrow < V_);

  v8f acc[8];
  #pragma unroll
  for (int t = 0; t < 8; ++t) acc[t] = (v8f){0.f, 0.f, 0.f, 0.f, 0.f, 0.f, 0.f, 0.f};

  for (int kt = 0; kt < H_; kt += 32) {
    stage_tile(lA, h1, 0, H_, kt);
    __syncthreads();
    BF16Frag Bf;
    if (valid) {
      const float* p = W2 + (long long)nrow * H_ + kt + khalf * 8;
      float4 x0 = *(const float4*)(p);
      float4 x1 = *(const float4*)(p + 4);
      float4 y0 = *(const float4*)(p + 16);
      float4 y1 = *(const float4*)(p + 20);
      Bf.v[0]  = (__bf16)x0.x; Bf.v[1]  = (__bf16)x0.y; Bf.v[2]  = (__bf16)x0.z; Bf.v[3]  = (__bf16)x0.w;
      Bf.v[4]  = (__bf16)x1.x; Bf.v[5]  = (__bf16)x1.y; Bf.v[6]  = (__bf16)x1.z; Bf.v[7]  = (__bf16)x1.w;
      Bf.v[8]  = (__bf16)y0.x; Bf.v[9]  = (__bf16)y0.y; Bf.v[10] = (__bf16)y0.z; Bf.v[11] = (__bf16)y0.w;
      Bf.v[12] = (__bf16)y1.x; Bf.v[13] = (__bf16)y1.y; Bf.v[14] = (__bf16)y1.z; Bf.v[15] = (__bf16)y1.w;
    } else {
      #pragma unroll
      for (int i = 0; i < 8; ++i) Bf.u[i] = 0u;
    }
    #pragma unroll
    for (int t = 0; t < 8; ++t) {
      BF16Frag Af;
      frag_from_lds_bf16(Af, lA, t * 16 + nl, khalf);
      acc[t] = __builtin_amdgcn_wmma_f32_16x16x32_bf16(
          false, Af.v, false, Bf.v, (short)0, acc[t], false, false);
    }
    __syncthreads();
  }
  if (valid) {
    float bias = b2[nrow];
    #pragma unroll
    for (int t = 0; t < 8; ++t)
      #pragma unroll
      for (int j = 0; j < 8; ++j) {
        int m = t * 16 + j + 8 * khalf;      // batch row
        logits[(long long)m * V_ + nrow] = acc[t][j] + bias;
      }
  }
}

// ---------------- vocab softmax stats + extended-vocab mix ----------------
__global__ __launch_bounds__(256) void k_rowstats(const float* __restrict__ logits,
    float* __restrict__ rowM, float* __restrict__ rowS) {
  __shared__ float sm[256];
  int b = blockIdx.x, tid = threadIdx.x;
  const float* l = logits + (long long)b * V_;
  float mx = -1e30f;
  for (int v = tid; v < V_; v += 256) mx = fmaxf(mx, l[v]);
  mx = block_max256(mx, sm);
  float sum = 0.f;
  for (int v = tid; v < V_; v += 256) sum += expf(l[v] - mx);
  sum = block_sum256(sum, sm);
  if (tid == 0) { rowM[b] = mx; rowS[b] = sum; }
}

__global__ __launch_bounds__(256) void k_vocab_mix(const float* __restrict__ logits,
    const float* __restrict__ rowM, const float* __restrict__ rowS,
    const float* __restrict__ p_gen, float* __restrict__ out0) {
  const long long NV = V_ + EXTRA_;
  long long idx = (long long)blockIdx.x * 256 + threadIdx.x;
  if (idx >= (long long)B_ * NV) return;
  int b = (int)(idx / NV);
  long long j = idx % NV;
  float v = 0.f;
  if (j < V_) v = p_gen[b] * expf(logits[(long long)b * V_ + j] - rowM[b]) / rowS[b];
  out0[idx] = v;
}

__global__ __launch_bounds__(256) void k_scatter(const int* __restrict__ ebev,
    const float* __restrict__ atten, const float* __restrict__ p_gen,
    float* __restrict__ out0) {
  int idx = blockIdx.x * 256 + threadIdx.x;   // B*S
  int b = idx / S_;
  int tgt = ebev[idx];
  atomicAdd(&out0[(long long)b * (V_ + EXTRA_) + tgt], (1.f - p_gen[b]) * atten[idx]);
}

__global__ __launch_bounds__(256) void k_cliplog(float* __restrict__ out0) {
  long long idx = (long long)blockIdx.x * 256 + threadIdx.x;
  if (idx >= (long long)B_ * (V_ + EXTRA_)) return;
  out0[idx] = logf(fmaxf(out0[idx], EPS_));
}

// ==================== launch ====================
extern "C" void kernel_launch(void* const* d_in, const int* in_sizes, int n_in,
                              void* d_out, int out_size, void* d_ws, size_t ws_size,
                              hipStream_t stream) {
  const int*   inputs     = (const int*)  d_in[0];
  const float* dec_hidden = (const float*)d_in[1];
  const float* enc        = (const float*)d_in[2];
  const float* mask       = (const float*)d_in[3];
  const float* context    = (const float*)d_in[4];
  const int*   ebev       = (const int*)  d_in[5];
  const float* coverage   = (const float*)d_in[7];
  const float* embed      = (const float*)d_in[8];
  const float* ln_g       = (const float*)d_in[9];
  const float* ln_b       = (const float*)d_in[10];
  const float* xc_W       = (const float*)d_in[11];
  const float* xc_b       = (const float*)d_in[12];
  const float* W_ih       = (const float*)d_in[13];
  const float* W_hh       = (const float*)d_in[14];
  const float* b_ih       = (const float*)d_in[15];
  const float* b_hh       = (const float*)d_in[16];
  const float* Wh         = (const float*)d_in[17];
  const float* Ws         = (const float*)d_in[18];
  const float* Ws_b       = (const float*)d_in[19];
  const float* Wc         = (const float*)d_in[20];
  const float* Vv         = (const float*)d_in[21];
  const float* pg_W       = (const float*)d_in[22];
  const float* pg_b       = (const float*)d_in[23];
  const float* out1_W     = (const float*)d_in[24];
  const float* out1_b     = (const float*)d_in[25];
  const float* out2_W     = (const float*)d_in[26];
  const float* out2_b     = (const float*)d_in[27];

  float* out       = (float*)d_out;
  float* out_final = out;                                          // [B, V+EXTRA]
  float* out_st    = out_final + (long long)B_ * (V_ + EXTRA_);    // [1,B,H]
  float* out_ht    = out_st + (long long)B_ * H_;                  // [B,H]
  float* out_covn  = out_ht + (long long)B_ * H_;                  // [B,S]
  float* out_covl  = out_covn + (long long)B_ * S_;                // [B]

  float* w = (float*)d_ws;
  float* xln    = w; w += (long long)B_ * E_;
  float* x      = w; w += (long long)B_ * E_;
  float* gx     = w; w += (long long)B_ * 3 * H_;
  float* gh     = w; w += (long long)B_ * 3 * H_;
  float* st     = w; w += (long long)B_ * H_;
  float* dfeat  = w; w += (long long)B_ * H_;
  float* e_t    = w; w += (long long)B_ * S_;
  float* atten  = w; w += (long long)B_ * S_;
  float* pgen   = w; w += B_;
  float* h1     = w; w += (long long)B_ * H_;
  float* logits = w; w += (long long)B_ * V_;
  float* rowM   = w; w += B_;
  float* rowS   = w; w += B_;

  hipMemsetAsync(e_t, 0, (size_t)B_ * S_ * sizeof(float), stream);

  // embedding + layernorm
  k_embed_ln<<<B_, 256, 0, stream>>>(inputs, embed, ln_g, ln_b, xln);
  // x = concat(xln, context) @ xc_W.T + xc_b
  k_dot_cat2<<<(B_ * E_) / 256, 256, 0, stream>>>(xln, context, xc_W, xc_b, x, E_, E_, H_);
  // GRU gates
  k_dot<<<(B_ * 3 * H_) / 256, 256, 0, stream>>>(x, W_ih, b_ih, gx, 3 * H_, E_);
  k_dot<<<(B_ * 3 * H_) / 256, 256, 0, stream>>>(dec_hidden, W_hh, b_hh, gh, 3 * H_, H_);
  k_gru<<<(B_ * H_) / 256, 256, 0, stream>>>(gx, gh, dec_hidden, st, out_st);
  // dec_feat = s_t @ Ws.T + Ws_b
  k_dot<<<(B_ * H_) / 256, 256, 0, stream>>>(st, Ws, Ws_b, dfeat, H_, H_);
  // fused additive attention scores (WMMA + TDM double-buffer)
  dim3 g5((B_ * S_) / 128, H_ / 128);
  k_attn_scores<<<g5, 256, 0, stream>>>(enc, Wh, dfeat, coverage, Wc, Vv, e_t);
  // softmax over S + coverage update / loss
  k_softmax_cov<<<B_, 256, 0, stream>>>(e_t, mask, coverage, atten, out_covn, out_covl);
  // context vector
  k_context<<<(B_ * H_) / 256, 256, 0, stream>>>(atten, enc, out_ht);
  // p_gen gate
  k_pgen<<<B_, 256, 0, stream>>>(out_ht, st, x, pg_W, pg_b, pgen);
  // h1 = concat(s_t, h_t) @ out1_W.T + out1_b
  k_dot_cat2<<<(B_ * H_) / 256, 256, 0, stream>>>(st, out_ht, out1_W, out1_b, h1, H_, H_, H_);
  // logits = h1 @ out2_W.T + out2_b (WMMA)
  k_out2_gemm<<<(V_ + 127) / 128, 256, 0, stream>>>(h1, out2_W, out2_b, logits);
  // softmax over V, mix with p_gen, scatter copy dist, clip+log
  k_rowstats<<<B_, 256, 0, stream>>>(logits, rowM, rowS);
  long long nfin = (long long)B_ * (V_ + EXTRA_);
  k_vocab_mix<<<(unsigned)((nfin + 255) / 256), 256, 0, stream>>>(logits, rowM, rowS, pgen, out_final);
  k_scatter<<<(B_ * S_) / 256, 256, 0, stream>>>(ebev, atten, pgen, out_final);
  k_cliplog<<<(unsigned)((nfin + 255) / 256), 256, 0, stream>>>(out_final);
}
